// baseLSNN_36996848288001
// MI455X (gfx1250) — matrix operand order repaired and checked by reading, not compiled, not run on hardware
//
#include <hip/hip_runtime.h>

// ---------------- LSNN problem constants ----------------
#define BATCH   128
#define TSTEPS  256
#define NIN     256
#define NREC    1024
#define KDIM    (NIN + NREC)       // 1280 : A = [x_t | z]
#define KCH     (KDIM / 32)        // 40 K-chunks of 32 for wmma f16 k=32
#define NTILES  (NREC / 16)        // 64 N tiles of 16
#define WAVES   16                 // waves per block
#define TPB     (WAVES * 32)       // 512 threads
#define TPW     (NTILES / WAVES)   // 4 N-tiles per wave
#define NBLOCKS (BATCH / 16)       // 8 blocks, 16 batch rows each

#define THR     0.01f
#define BETA    1.8f
#define REFP    2.0f
#define DECAY_V 0.9512294245007140f   // exp(-1/20)
#define DECAY_A 0.9512294245007140f   // exp(-1/20)

// ---------------- LDS layout (dynamic, ~297 KB) ----------------
#define XSTR  264                              // halves per x-panel row (256 + pad)
#define ZSTR  1032                             // halves per z row (1024 + pad)
#define SCH   28                               // floats per (tile,lane) state chunk: v8,a8,lsd8 + 4 pad
#define XBYTES (16 * XSTR * 2)                 // 8448
#define ZBYTES (2 * 16 * ZSTR * 2)             // 66048 (double buffered z)
#define SBYTES (NTILES * 32 * SCH * 4)         // 229376 (lane-major interleaved states)
#define SMEM_BYTES (XBYTES + ZBYTES + SBYTES)  // 303872 <= 320KB

typedef _Float16 h8  __attribute__((ext_vector_type(8)));
typedef _Float16 h16 __attribute__((ext_vector_type(16)));
typedef float    vf8 __attribute__((ext_vector_type(8)));
typedef float    vf4 __attribute__((ext_vector_type(4)));

// Packed f16 weights in WMMA-B lane-major layout:
// wp[((tileN*KCH + kc)*32 + lane)*16 + h]
// element (k, n): tileN=n>>4, kc=k>>5, lane=(n&15)+16*((k>>3)&1), h=(k&7)+8*((k&31)>>4)
__device__ _Float16 g_wp[KDIM * NREC];

// ---------------- one-time weight repack (+ diagonal mask, fp32 -> f16) ----------------
__global__ void lsnn_repack_kernel(const float* __restrict__ w_in,
                                   const float* __restrict__ w_rec) {
  int idx = blockIdx.x * blockDim.x + threadIdx.x;
  if (idx >= KDIM * NREC) return;
  int k = idx / NREC;
  int n = idx - k * NREC;
  float val;
  if (k < NIN) {
    val = w_in[k * NREC + n];
  } else {
    int kr = k - NIN;
    val = (kr == n) ? 0.0f : w_rec[kr * NREC + n];   // mask self-connections
  }
  int tn = n >> 4, nn = n & 15;
  int kc = k >> 5, kk = k & 31;
  int lane = nn + 16 * ((kk >> 3) & 1);
  int h    = (kk & 7) + 8 * (kk >> 4);
  g_wp[(((size_t)tn * KCH + kc) * 32 + lane) * 16 + h] = (_Float16)val;
}

// ---------------- persistent per-batch-tile LSNN scan ----------------
__global__ __launch_bounds__(TPB, 1)
void lsnn_scan_kernel(const float* __restrict__ x, float* __restrict__ out) {
  extern __shared__ char smem[];
  _Float16* xs = (_Float16*)smem;                        // [16][XSTR] halves
  _Float16* zs = (_Float16*)(smem + XBYTES);             // [2][16][ZSTR] halves
  float*    ss = (float*)(smem + XBYTES + ZBYTES);       // [NTILES][32][SCH] lane-major states

  const int tid  = threadIdx.x;
  const int lane = tid & 31;
  const int wave = tid >> 5;
  const int b0   = blockIdx.x * 16;

  // A-operand lane mapping (16-bit A 16x32, ISA 7.12.2):
  // lane L: row m=L&15; halves 0..7 -> K = 8*(L>>4)+0..7 ; halves 8..15 -> +16
  const int am  = lane & 15;
  const int ahi = lane >> 4;

  // zero initial state (z=v=a=lsd=0)
  for (int i = tid; i < 2 * 16 * ZSTR; i += TPB) zs[i] = (_Float16)0.0f;
  for (int i = tid; i < NTILES * 32 * SCH; i += TPB) ss[i] = 0.0f;
  __syncthreads();

  for (int t = 0; t < TSTEPS; ++t) {
    // ---- stage x_t into LDS as f16: row = wave, cols = lane*8..lane*8+7 ----
    // vectorized: 2x global_load_b128 -> packed cvt -> 1x ds_store_b128
    {
      const float* src = x + (((size_t)(b0 + wave)) * TSTEPS + t) * NIN + lane * 8;
      vf4 s0 = *(const vf4*)src;
      vf4 s1 = *(const vf4*)(src + 4);
      h8 hv;
#pragma unroll
      for (int j = 0; j < 4; ++j) { hv[j] = (_Float16)s0[j]; hv[4 + j] = (_Float16)s1[j]; }
      *(h8*)(xs + wave * XSTR + lane * 8) = hv;
    }
    const int pr = t & 1;        // z buffer holding z_{t-1}
    const int pw = pr ^ 1;       // z buffer receiving z_t
    __syncthreads();             // x_t + z_{t-1} visible to all waves

    // ---- i_in = [x_t | z_{t-1}] @ W  via WMMA f16->f32 ----
    vf8 acc[TPW];
#pragma unroll
    for (int i = 0; i < TPW; ++i) acc[i] = (vf8){0,0,0,0,0,0,0,0};

    // K-chunks 0..7 read the x panel
    for (int kc = 0; kc < NIN / 32; ++kc) {
      union { h16 v; h8 h[2]; } A;
      const _Float16* ap = xs + am * XSTR + kc * 32 + ahi * 8;
      A.h[0] = *(const h8*)ap;
      A.h[1] = *(const h8*)(ap + 16);
#pragma unroll
      for (int ti = 0; ti < TPW; ++ti) {
        const int tn = wave * TPW + ti;
        const _Float16* bp = g_wp + (((size_t)tn * KCH + kc) * 32 + lane) * 16;
        h16 B = *(const h16*)bp;
        acc[ti] = __builtin_amdgcn_wmma_f32_16x16x32_f16(
            false, A.v, false, B, (short)0, acc[ti], false, false);
      }
    }
    // K-chunks 8..39 read the z_{t-1} panel
    for (int kc = NIN / 32; kc < KCH; ++kc) {
      union { h16 v; h8 h[2]; } A;
      const _Float16* ap = zs + (size_t)(pr * 16 + am) * ZSTR + (kc * 32 - NIN) + ahi * 8;
      A.h[0] = *(const h8*)ap;
      A.h[1] = *(const h8*)(ap + 16);
#pragma unroll
      for (int ti = 0; ti < TPW; ++ti) {
        const int tn = wave * TPW + ti;
        const _Float16* bp = g_wp + (((size_t)tn * KCH + kc) * 32 + lane) * 16;
        h16 B = *(const h16*)bp;
        acc[ti] = __builtin_amdgcn_wmma_f32_16x16x32_f16(
            false, A.v, false, B, (short)0, acc[ti], false, false);
      }
    }

    // ---- fused ALIF / refractory update, write z_t ----
    // C/D layout: lane L, vgpr r -> M = r + 8*(L>>4), N = (L&15) + 16*tileN
    // States are lane-private -> lane-major LDS chunks, b128 vector access.
#pragma unroll
    for (int ti = 0; ti < TPW; ++ti) {
      const int tn = wave * TPW + ti;
      const int n  = tn * 16 + (lane & 15);
      float* base = ss + ((size_t)tn * 32 + lane) * SCH;
      float vv[8], aa[8], ll[8];
      *(vf4*)&vv[0] = *(const vf4*)(base + 0);
      *(vf4*)&vv[4] = *(const vf4*)(base + 4);
      *(vf4*)&aa[0] = *(const vf4*)(base + 8);
      *(vf4*)&aa[4] = *(const vf4*)(base + 12);
      *(vf4*)&ll[0] = *(const vf4*)(base + 16);
      *(vf4*)&ll[4] = *(const vf4*)(base + 20);
#pragma unroll
      for (int r = 0; r < 8; ++r) {
        const int m = r + 8 * ahi;
        const float zprev = (float)zs[(size_t)(pr * 16 + m) * ZSTR + n];
        const float na   = DECAY_A * aa[r] + (1.0f - DECAY_A) * zprev;
        const float athr = THR + BETA * na;
        const float nv   = DECAY_V * vv[r] + acc[ti][r] - athr * zprev;
        const float vsc  = nv - athr;
        float zn = (vsc > 0.0f) ? 1.0f : 0.0f;
        const float spike  = zn;                          // pre-refractory spike
        const float nonref = (ll[r] >= REFP) ? 1.0f : 0.0f;
        zn *= nonref;
        const float nlsd = (ll[r] + 1.0f) * (1.0f - spike);
        vv[r] = nv; aa[r] = na; ll[r] = nlsd;
        zs[(size_t)(pw * 16 + m) * ZSTR + n] = (_Float16)zn;
        out[(((size_t)(b0 + m)) * TSTEPS + t) * NREC + n] = zn;
      }
      *(vf4*)(base + 0)  = *(const vf4*)&vv[0];
      *(vf4*)(base + 4)  = *(const vf4*)&vv[4];
      *(vf4*)(base + 8)  = *(const vf4*)&aa[0];
      *(vf4*)(base + 12) = *(const vf4*)&aa[4];
      *(vf4*)(base + 16) = *(const vf4*)&ll[0];
      *(vf4*)(base + 20) = *(const vf4*)&ll[4];
    }
    __syncthreads();   // GEMM reads + z_t writes done before next iter's overwrite
  }
}

extern "C" void kernel_launch(void* const* d_in, const int* in_sizes, int n_in,
                              void* d_out, int out_size, void* d_ws, size_t ws_size,
                              hipStream_t stream) {
  (void)in_sizes; (void)n_in; (void)out_size; (void)d_ws; (void)ws_size;
  const float* x     = (const float*)d_in[0];
  const float* w_in  = (const float*)d_in[1];
  const float* w_rec = (const float*)d_in[2];
  float* out = (float*)d_out;

  // 1) repack weights (mask diag, fp32->f16, WMMA-B tiled layout) into g_wp
  const int total = KDIM * NREC;
  lsnn_repack_kernel<<<(total + 255) / 256, 256, 0, stream>>>(w_in, w_rec);

  // 2) persistent scan: 8 blocks x 512 threads, ~297KB dynamic LDS each
  lsnn_scan_kernel<<<NBLOCKS, TPB, SMEM_BYTES, stream>>>(x, out);
}
